// hierarchical_cluster_mil_53661321396720
// MI455X (gfx1250) — matrix operand order repaired and checked by reading, not compiled, not run on hardware
//
#include <hip/hip_runtime.h>

#define B_ 16
#define P_ 4096
#define F_ 1024
#define Z_ 256
#define C_ 10
#define NOUT_ 2
#define BN_EPS_ 1e-5f

typedef __attribute__((ext_vector_type(16))) __bf16 v16bf;
typedef __attribute__((ext_vector_type(8)))  float  v8f;
typedef __attribute__((ext_vector_type(16))) unsigned short v16u;
typedef __attribute__((ext_vector_type(8)))  unsigned short v8u;

union VU { v16u u; v16bf b; };

// Native f32 -> bf16 (compiler lowers to v_cvt_*bf16_f32 on gfx1250).
__device__ __forceinline__ unsigned short f2bf(float f) {
  union { __bf16 b; unsigned short u; } cv;
  cv.b = (__bf16)f;
  return cv.u;
}
__device__ __forceinline__ float bf2f(unsigned short h) {
  union { unsigned u; float f; } x; x.u = ((unsigned)h) << 16;
  return x.f;
}

__device__ __forceinline__ void wait_async0() {
#if __has_builtin(__builtin_amdgcn_s_wait_asynccnt)
  __builtin_amdgcn_s_wait_asynccnt(0);
#else
  asm volatile("s_wait_asynccnt 0" ::: "memory");
#endif
}

// Per-lane 16-byte global -> LDS async copy (gfx1250 GLOBAL_LOAD_ASYNC_TO_LDS_B128).
__device__ __forceinline__ void async_g2l_b128(unsigned lds_off, const void* gptr) {
  asm volatile("global_load_async_to_lds_b128 %0, %1, off"
               :: "v"(lds_off), "v"((unsigned long long)(uintptr_t)gptr)
               : "memory");
}

// ---------------------------------------------------------------------------
// Pack a row-major fp32 [K,N] weight into bf16 in the exact per-lane register
// layout of the v_wmma_f32_16x16x32_bf16 B operand:
//   tile (kt, nt): lane l (<16): col n = nt*16 + l,   K = kt*32 + {0..7,16..23}[e]
//                  lane l (>=16): col n = nt*16+l-16, K = kt*32 + {8..15,24..31}[e]
// ---------------------------------------------------------------------------
__global__ __launch_bounds__(256) void pack_b_kernel(
    const float* __restrict__ W, unsigned short* __restrict__ out, int K, int N) {
  int o = blockIdx.x * blockDim.x + threadIdx.x;
  if (o >= K * N) return;
  int e    = o & 15;
  int lane = (o >> 4) & 31;
  int t2   = o >> 9;
  int NT   = N >> 4;
  int nt   = t2 % NT;
  int kt   = t2 / NT;
  int n = nt * 16 + (lane & 15);
  int k = kt * 32 + (e & 7) + ((e >> 3) << 4) + ((lane >= 16) ? 8 : 0);
  out[o] = f2bf(W[k * N + n]);
}

// A-operand packed layout for v_wmma_f32_16x16x32_bf16:
// lane<16 holds K {0..7,16..23}; lane>=16 holds K {8..15,24..31} of row m.
// 8 consecutive K starting at k0 in {0,8,16,24} land contiguously in one lane.

// ---------------------------------------------------------------------------
// Encoder: emb = relu(bags @ enc_W + enc_b)   [65536,1024]x[1024,256] -> bf16
// Block: 64 rows x 256 cols, 8 waves; wave owns 32 cols; 8 WMMA per K-step.
// LDS double-buffered: one barrier per K-step; next A tile pipelined in regs.
// ---------------------------------------------------------------------------
__global__ __launch_bounds__(256) void enc_gemm_kernel(
    const float* __restrict__ A, const unsigned short* __restrict__ Bp,
    const float* __restrict__ bias, unsigned short* __restrict__ outBf) {
  __shared__ __align__(32) unsigned short lA[2][4 * 512];  // 2 x 4KB packed tiles
  const int row0 = blockIdx.x * 64;
  const int tid = threadIdx.x;
  const int wave = tid >> 5, lane = tid & 31;
  const int nt0 = wave * 2, nt1 = nt0 + 1;
  const int NT = Z_ / 16;
  // staging: thread covers row m_st, K chunk [k0, k0+8)
  const int m_st = tid >> 2;
  const int k0   = (tid & 3) << 3;
  const int stTile = m_st >> 4;
  const int stLane = (m_st & 15) + (((k0 >> 3) & 1) << 4);
  const int stE    = (k0 >> 4) << 3;
  const int stIdx  = stTile * 512 + stLane * 16 + stE;     // 16B aligned slot
  const float* srcBase = A + (size_t)(row0 + m_st) * F_ + k0;
  v8f acc[4][2] = {};
  float rg[8];
#pragma unroll
  for (int i = 0; i < 8; ++i) rg[i] = srcBase[i];          // prologue: tile 0
  for (int kt = 0; kt < F_ / 32; ++kt) {
    const int cur = kt & 1;
    {
      v8u hv;
#pragma unroll
      for (int i = 0; i < 8; ++i) hv[i] = f2bf(rg[i]);
      *reinterpret_cast<v8u*>(&lA[cur][stIdx]) = hv;
    }
    if (kt + 1 < F_ / 32) {
      const float* src = srcBase + ((kt + 1) << 5);
#pragma unroll
      for (int i = 0; i < 8; ++i) rg[i] = src[i];          // pipeline next tile
      if (kt + 2 < F_ / 32) __builtin_prefetch(src + 32, 0, 3);
    }
    __syncthreads();
    VU bv0, bv1;
    bv0.u = *reinterpret_cast<const v16u*>(Bp + (((size_t)(kt * NT + nt0) * 32 + lane) << 4));
    bv1.u = *reinterpret_cast<const v16u*>(Bp + (((size_t)(kt * NT + nt1) * 32 + lane) << 4));
#pragma unroll
    for (int t = 0; t < 4; ++t) {
      VU av; av.u = *reinterpret_cast<const v16u*>(&lA[cur][t * 512 + lane * 16]);
      acc[t][0] = __builtin_amdgcn_wmma_f32_16x16x32_bf16(false, av.b, false, bv0.b, (short)0, acc[t][0], false, false);
      acc[t][1] = __builtin_amdgcn_wmma_f32_16x16x32_bf16(false, av.b, false, bv1.b, (short)0, acc[t][1], false, false);
    }
  }
  const int n0 = wave * 32 + (lane & 15);
  const int n1 = n0 + 16;
  const float b0 = bias[n0], b1 = bias[n1];
#pragma unroll
  for (int t = 0; t < 4; ++t) {
    const int mb = row0 + t * 16 + ((lane >= 16) ? 8 : 0);
#pragma unroll
    for (int r = 0; r < 8; ++r) {
      float x0 = acc[t][0][r] + b0; x0 = x0 > 0.f ? x0 : 0.f;
      float x1 = acc[t][1][r] + b1; x1 = x1 > 0.f ? x1 : 0.f;
      outBf[(size_t)(mb + r) * Z_ + n0] = f2bf(x0);
      outBf[(size_t)(mb + r) * Z_ + n1] = f2bf(x1);
    }
  }
}

// ---------------------------------------------------------------------------
// Region attention score: s[m] = tanh(emb[m,:] @ ra_W1 + ra_b1) . ra_w2 + ra_b2
// 64 rows/block; A staged via GLOBAL_LOAD_ASYNC_TO_LDS_B128, double-buffered:
// the async DMA for tile kt+1 overlaps the 8 WMMAs of tile kt.
// ---------------------------------------------------------------------------
__global__ __launch_bounds__(256) void attn_score_kernel(
    const unsigned short* __restrict__ embBf, const unsigned short* __restrict__ Bp,
    const float* __restrict__ b1v, const float* __restrict__ w2,
    const float* __restrict__ b2, float* __restrict__ s) {
  __shared__ __align__(32) unsigned short lA[2][4 * 512];
  __shared__ float sred[64];
  const int row0 = blockIdx.x * 64;
  const int tid = threadIdx.x;
  const int wave = tid >> 5, lane = tid & 31;
  const int nt0 = wave * 2, nt1 = nt0 + 1;
  const int NT = Z_ / 16;
  const int m_st = tid >> 2;
  const int k0   = (tid & 3) << 3;
  const int stTile = m_st >> 4;
  const int stLane = (m_st & 15) + (((k0 >> 3) & 1) << 4);
  const int stE    = (k0 >> 4) << 3;
  const unsigned lds_off =
      (unsigned)(uintptr_t)&lA[0][stTile * 512 + stLane * 16 + stE];
  const unsigned short* gsrc = embBf + (size_t)(row0 + m_st) * Z_ + k0;
  if (tid < 64) sred[tid] = 0.f;
  v8f acc[4][2] = {};
  async_g2l_b128(lds_off, gsrc);                         // prologue: tile 0
  for (int kt = 0; kt < Z_ / 32; ++kt) {
    const int cur = kt & 1;
    wait_async0();
    __syncthreads();
    if (kt + 1 < Z_ / 32)                                // DMA next tile now,
      async_g2l_b128(lds_off + ((kt + 1) & 1) * 4096,    // overlaps WMMAs below
                     gsrc + ((kt + 1) << 5));
    VU bv0, bv1;
    bv0.u = *reinterpret_cast<const v16u*>(Bp + (((size_t)(kt * NT + nt0) * 32 + lane) << 4));
    bv1.u = *reinterpret_cast<const v16u*>(Bp + (((size_t)(kt * NT + nt1) * 32 + lane) << 4));
#pragma unroll
    for (int t = 0; t < 4; ++t) {
      VU av; av.u = *reinterpret_cast<const v16u*>(&lA[cur][t * 512 + lane * 16]);
      acc[t][0] = __builtin_amdgcn_wmma_f32_16x16x32_bf16(false, av.b, false, bv0.b, (short)0, acc[t][0], false, false);
      acc[t][1] = __builtin_amdgcn_wmma_f32_16x16x32_bf16(false, av.b, false, bv1.b, (short)0, acc[t][1], false, false);
    }
  }
  const int n0 = wave * 32 + (lane & 15);
  const int n1 = n0 + 16;
  const float bb0 = b1v[n0], bb1 = b1v[n1];
  const float ww0 = w2[n0],  ww1 = w2[n1];
#pragma unroll
  for (int t = 0; t < 4; ++t) {
#pragma unroll
    for (int r = 0; r < 8; ++r) {
      float v = tanhf(acc[t][0][r] + bb0) * ww0 + tanhf(acc[t][1][r] + bb1) * ww1;
      for (int off = 1; off < 16; off <<= 1) v += __shfl_xor(v, off, 32);
      if (lane == 0)  atomicAdd(&sred[t * 16 + r], v);       // ds_add_f32
      if (lane == 16) atomicAdd(&sred[t * 16 + 8 + r], v);
    }
  }
  __syncthreads();
  if (tid < 64) s[row0 + tid] = sred[tid] + b2[0];
}

// ---------------------------------------------------------------------------
// Per-bag segmented softmax over clusters + weighted segment-sum of emb.
// ---------------------------------------------------------------------------
__global__ __launch_bounds__(256) void segment_kernel(
    const float* __restrict__ s, const int* __restrict__ clusters,
    const unsigned short* __restrict__ embBf, float* __restrict__ reg) {
  __shared__ float ex[P_];
  __shared__ int   cl[P_];
  __shared__ float red[C_ * 256];
  __shared__ float smax[C_];
  __shared__ float rden[C_];
  const int b = blockIdx.x;
  const int tid = threadIdx.x;
  for (int p = tid; p < P_; p += 256) cl[p] = clusters[b * P_ + p];
  float mx[C_];
#pragma unroll
  for (int c = 0; c < C_; ++c) mx[c] = -3.4e38f;
  __syncthreads();
  for (int p = tid; p < P_; p += 256) {
    float v = s[b * P_ + p];
    int c = cl[p];
#pragma unroll
    for (int i = 0; i < C_; ++i) mx[i] = (c == i && v > mx[i]) ? v : mx[i];
    ex[p] = v;
  }
#pragma unroll
  for (int c = 0; c < C_; ++c) red[c * 256 + tid] = mx[c];
  __syncthreads();
  for (int off = 128; off > 0; off >>= 1) {
    if (tid < off) {
#pragma unroll
      for (int c = 0; c < C_; ++c) {
        float a = red[c * 256 + tid], bv = red[c * 256 + tid + off];
        red[c * 256 + tid] = a > bv ? a : bv;
      }
    }
    __syncthreads();
  }
  if (tid < C_) smax[tid] = red[tid * 256];
  __syncthreads();
  float sm[C_];
#pragma unroll
  for (int c = 0; c < C_; ++c) sm[c] = 0.f;
  for (int p = tid; p < P_; p += 256) {
    int c = cl[p];
    float e = __expf(ex[p] - smax[c]);
    ex[p] = e;
#pragma unroll
    for (int i = 0; i < C_; ++i) sm[i] += (c == i) ? e : 0.f;
  }
#pragma unroll
  for (int c = 0; c < C_; ++c) red[c * 256 + tid] = sm[c];
  __syncthreads();
  for (int off = 128; off > 0; off >>= 1) {
    if (tid < off) {
#pragma unroll
      for (int c = 0; c < C_; ++c) red[c * 256 + tid] += red[c * 256 + tid + off];
    }
    __syncthreads();
  }
  if (tid < C_) rden[tid] = 1.f / red[tid * 256];
  __syncthreads();
  float acc[C_];
#pragma unroll
  for (int c = 0; c < C_; ++c) acc[c] = 0.f;
  const unsigned short* eb = embBf + (size_t)b * P_ * Z_ + tid;
  for (int p = 0; p < P_; ++p) {
    int c = cl[p];
    float w = ex[p];
    float e = bf2f(eb[(size_t)p * Z_]);
#pragma unroll
    for (int i = 0; i < C_; ++i) acc[i] += ((c == i) ? w : 0.f) * e;
  }
#pragma unroll
  for (int c = 0; c < C_; ++c)
    reg[((size_t)b * C_ + c) * Z_ + tid] = acc[c] * rden[c];
}

// ---------------------------------------------------------------------------
// BN + region head + slide attention + output head. One block per bag.
// ---------------------------------------------------------------------------
__global__ __launch_bounds__(256) void head_kernel(
    const float* __restrict__ reg, const float* g, const float* be,
    const float* mean, const float* var, const float* rhW, const float* rhb,
    const float* saW1, const float* sab1, const float* saw2, const float* sab2,
    const float* shW, const float* shb, float* __restrict__ out) {
  __shared__ float r1[C_][Z_];
  __shared__ float r2[C_][Z_];
  __shared__ float red[Z_];
  __shared__ float aw[C_];
  const int b = blockIdx.x, tid = threadIdx.x;
  const float scale = rsqrtf(var[tid] + BN_EPS_) * g[tid];
  const float shift = be[tid] - mean[tid] * scale;
#pragma unroll
  for (int c = 0; c < C_; ++c)
    r1[c][tid] = reg[((size_t)b * C_ + c) * Z_ + tid] * scale + shift;
  __syncthreads();
  float acc[C_];
#pragma unroll
  for (int c = 0; c < C_; ++c) acc[c] = rhb[tid];
  for (int k = 0; k < Z_; ++k) {
    float w = rhW[k * Z_ + tid];
#pragma unroll
    for (int c = 0; c < C_; ++c) acc[c] += r1[c][k] * w;
  }
#pragma unroll
  for (int c = 0; c < C_; ++c) r2[c][tid] = acc[c];
  __syncthreads();
#pragma unroll
  for (int c = 0; c < C_; ++c) acc[c] = sab1[tid];
  for (int k = 0; k < Z_; ++k) {
    float w = saW1[k * Z_ + tid];
#pragma unroll
    for (int c = 0; c < C_; ++c) acc[c] += r2[c][k] * w;
  }
  const float w2v = saw2[tid];
  for (int c = 0; c < C_; ++c) {
    red[tid] = tanhf(acc[c]) * w2v;
    __syncthreads();
    for (int off = 128; off > 0; off >>= 1) {
      if (tid < off) red[tid] += red[tid + off];
      __syncthreads();
    }
    if (tid == 0) aw[c] = red[0] + sab2[0];
    __syncthreads();
  }
  if (tid == 0) {
    float m = aw[0];
    for (int c = 1; c < C_; ++c) m = aw[c] > m ? aw[c] : m;
    float sum = 0.f;
    for (int c = 0; c < C_; ++c) { aw[c] = __expf(aw[c] - m); sum += aw[c]; }
    float inv = 1.f / sum;
    for (int c = 0; c < C_; ++c) aw[c] *= inv;
  }
  __syncthreads();
  float sl = 0.f;
#pragma unroll
  for (int c = 0; c < C_; ++c) sl += aw[c] * r2[c][tid];
  for (int no = 0; no < NOUT_; ++no) {
    red[tid] = sl * shW[tid * NOUT_ + no];
    __syncthreads();
    for (int off = 128; off > 0; off >>= 1) {
      if (tid < off) red[tid] += red[tid + off];
      __syncthreads();
    }
    if (tid == 0) out[b * NOUT_ + no] = red[0] + shb[no];
    __syncthreads();
  }
}

extern "C" void kernel_launch(void* const* d_in, const int* in_sizes, int n_in,
                              void* d_out, int out_size, void* d_ws, size_t ws_size,
                              hipStream_t stream) {
  const float* bags     = (const float*)d_in[0];
  const int*   clusters = (const int*)d_in[1];
  const float* enc_W    = (const float*)d_in[2];
  const float* enc_b    = (const float*)d_in[3];
  const float* ra_W1    = (const float*)d_in[4];
  const float* ra_b1    = (const float*)d_in[5];
  const float* ra_w2    = (const float*)d_in[6];
  const float* ra_b2    = (const float*)d_in[7];
  const float* bn_gamma = (const float*)d_in[8];
  const float* bn_beta  = (const float*)d_in[9];
  const float* bn_mean  = (const float*)d_in[10];
  const float* bn_var   = (const float*)d_in[11];
  const float* rh_W     = (const float*)d_in[12];
  const float* rh_b     = (const float*)d_in[13];
  const float* sa_W1    = (const float*)d_in[14];
  const float* sa_b1    = (const float*)d_in[15];
  const float* sa_w2    = (const float*)d_in[16];
  const float* sa_b2    = (const float*)d_in[17];
  const float* sh_W     = (const float*)d_in[18];
  const float* sh_b     = (const float*)d_in[19];
  float* out = (float*)d_out;
  (void)in_sizes; (void)n_in; (void)out_size; (void)ws_size;

  char* ws = (char*)d_ws;
  size_t off = 0;
  auto wsalloc = [&](size_t bytes) {
    char* p = ws + off;
    off = (off + bytes + 255) & ~(size_t)255;
    return p;
  };
  unsigned short* embBf = (unsigned short*)wsalloc((size_t)B_ * P_ * Z_ * 2);  // 32 MB
  unsigned short* encWp = (unsigned short*)wsalloc((size_t)F_ * Z_ * 2);       // 512 KB
  unsigned short* raW1p = (unsigned short*)wsalloc((size_t)Z_ * Z_ * 2);       // 128 KB
  float* sArr   = (float*)wsalloc((size_t)B_ * P_ * 4);                        // 256 KB
  float* regArr = (float*)wsalloc((size_t)B_ * C_ * Z_ * 4);                   // 160 KB

  pack_b_kernel<<<(F_ * Z_ + 255) / 256, 256, 0, stream>>>(enc_W, encWp, F_, Z_);
  pack_b_kernel<<<(Z_ * Z_ + 255) / 256, 256, 0, stream>>>(ra_W1, raW1p, Z_, Z_);
  enc_gemm_kernel<<<(B_ * P_) / 64, 256, 0, stream>>>(bags, encWp, enc_b, embBf);
  attn_score_kernel<<<(B_ * P_) / 64, 256, 0, stream>>>(embBf, raW1p, ra_b1, ra_w2, ra_b2, sArr);
  segment_kernel<<<B_, 256, 0, stream>>>(sArr, clusters, embBf, regArr);
  head_kernel<<<B_, 256, 0, stream>>>(regArr, bn_gamma, bn_beta, bn_mean, bn_var,
                                      rh_W, rh_b, sa_W1, sa_b1, sa_w2, sa_b2,
                                      sh_W, sh_b, out);
}
